// GNNHiddenStatePredictor_7198365188339
// MI455X (gfx1250) — matrix operand order repaired and checked by reading, not compiled, not run on hardware
//
#include <hip/hip_runtime.h>

// GNNHiddenStatePredictor — fused WMMA implementation for gfx1250 (MI455X).
//
// Exact algebraic reduction (same math as reference):
//  * out only consumes n[:, V:]  -> only agg rows i >= V are needed.
//  * h_full[i>=V] == 0           -> a[i] == 0 for those rows, so
//    e[i,j] = relu(b[j] + b_e1) is independent of i, and
//    agg[b,i,:] = adj[b,i,:] @ F[b]  with F[b,j] = relu(relu(b[j]+b_e1)@w_e2+b_e2).
//
// Data-movement strategy:
//  * All weights pre-transposed to f16 (Wt[col][k]) once per launch ->
//    every WMMA B-fragment is two contiguous global_load_b128 per lane.
//  * adj hidden rows pre-converted to f16 once -> A-fragments in the K=512
//    aggregation loop are two b128 loads per lane, no cvt in the hot loop.
//  * F is produced transposed in f16 (Ft[b][h][j]) so the aggregation GEMM
//    streams b128 B-fragments too.
//  * LDS is used only to pass tiles between waves inside the MLP chains.

#define B_   4
#define V_   384
#define H_   128
#define N_   512
#define HD_  64
#define TD_  128

typedef __attribute__((ext_vector_type(16))) _Float16 v16h;
typedef __attribute__((ext_vector_type(8)))  float    v8f;

__device__ __forceinline__ v8f wmma_f16(v16h a, v16h b, v8f c) {
  return __builtin_amdgcn_wmma_f32_16x16x32_f16(false, a, false, b, (short)0, c,
                                                false, false);
}

// ---- A-matrix fragment (16x32 f16) ----------------------------------------
// Per ISA 7.12.2: lane<16: row M=lane, elems 0..7 = K0..7, 8..15 = K16..23;
// lane>=16: row M=lane-16, elems 0..7 = K8..15, 8..15 = K24..31.
// Source: row-major f16 (LDS or global), row stride lda halfs.
__device__ __forceinline__ v16h a_frag_h16(const _Float16* base, int lda,
                                           int k0) {
  int lane = threadIdx.x & 31;
  const _Float16* p = base + (lane & 15) * lda + k0 + ((lane >> 4) << 3);
  union { v16h v; float4 q[2]; } u;
  u.q[0] = *(const float4*)(p);        // K k0+hf*8    .. +7
  u.q[1] = *(const float4*)(p + 16);   // K k0+hf*8+16 .. +23
  return u.v;
}

// From row-major f32 global with row stride lda (4x global_load_b128 + cvt).
__device__ __forceinline__ v16h a_frag_g32(const float* __restrict__ A, int lda,
                                           int k0) {
  int lane = threadIdx.x & 31;
  const float* p = A + (size_t)(lane & 15) * lda + k0 + ((lane >> 4) << 3);
  float4 x0 = *(const float4*)(p);
  float4 x1 = *(const float4*)(p + 4);
  float4 x2 = *(const float4*)(p + 16);
  float4 x3 = *(const float4*)(p + 20);
  v16h a;
  a[0]  = (_Float16)x0.x; a[1]  = (_Float16)x0.y;
  a[2]  = (_Float16)x0.z; a[3]  = (_Float16)x0.w;
  a[4]  = (_Float16)x1.x; a[5]  = (_Float16)x1.y;
  a[6]  = (_Float16)x1.z; a[7]  = (_Float16)x1.w;
  a[8]  = (_Float16)x2.x; a[9]  = (_Float16)x2.y;
  a[10] = (_Float16)x2.z; a[11] = (_Float16)x2.w;
  a[12] = (_Float16)x3.x; a[13] = (_Float16)x3.y;
  a[14] = (_Float16)x3.z; a[15] = (_Float16)x3.w;
  return a;
}

// ---- B-matrix fragment (32x16 f16) ----------------------------------------
// lane<16: column col0+lane, elems 0..7 -> K=k0+0..7, 8..15 -> K=k0+16..23;
// lane>=16: same col, elems 0..7 -> K=k0+8..15, 8..15 -> K=k0+24..31.
// Source is pre-transposed f16: Wt[col][k], row stride ldk -> per lane the 16
// K-values form two contiguous runs of 8, 16 apart: two global_load_b128.
__device__ __forceinline__ v16h b_frag_t(const _Float16* __restrict__ Wt,
                                         int ldk, int k0, int col0) {
  int lane = threadIdx.x & 31;
  const _Float16* p =
      Wt + (size_t)(col0 + (lane & 15)) * ldk + k0 + ((lane >> 4) << 3);
  union { v16h v; float4 q[2]; } u;
  u.q[0] = *(const float4*)(p);
  u.q[1] = *(const float4*)(p + 16);
  return u.v;
}

// ---- D tile helpers (16x16 f32): VGPR r, lane<16 -> row r, col lane; ------
// lane>=16 -> row r+8, col lane-16.
__device__ __forceinline__ v8f bias_relu(v8f d, const float* __restrict__ bias,
                                         int col0) {
  int lane = threadIdx.x & 31;
  float bv = bias[col0 + (lane & 15)];
#pragma unroll
  for (int r = 0; r < 8; ++r) d[r] = fmaxf(d[r] + bv, 0.f);
  return d;
}

template <int S>
__device__ __forceinline__ void store_tile_lds(_Float16* base, int col0, v8f d) {
  int lane = threadIdx.x & 31;
  int col = col0 + (lane & 15);
  int r0  = (lane >> 4) * 8;
#pragma unroll
  for (int r = 0; r < 8; ++r) base[(r0 + r) * S + col] = (_Float16)d[r];
}

__device__ __forceinline__ void store_tile_g(float* __restrict__ out, int ld,
                                             int col0, v8f d) {
  int lane = threadIdx.x & 31;
  int col = col0 + (lane & 15);
  int r0  = (lane >> 4) * 8;
#pragma unroll
  for (int r = 0; r < 8; ++r) out[(r0 + r) * ld + col] = d[r];
}

// ---------------------------------------------------------------------------
// Prep A: transpose+convert one f32 weight (K rows x C cols) to f16 Wt[C][K].
// ---------------------------------------------------------------------------
__global__ __launch_bounds__(256) void k_wt(const float* __restrict__ W,
                                            _Float16* __restrict__ Wt, int K,
                                            int C) {
  int idx = blockIdx.x * 256 + threadIdx.x;
  if (idx >= K * C) return;
  int k = idx / C, c = idx - k * C;
  Wt[(size_t)c * K + k] = (_Float16)W[idx];
}

// ---------------------------------------------------------------------------
// Prep B: convert hidden-row block of adj to f16: adjh[b*H+i'][j], i' = i-V.
// Vectorized 4-wide: b128 load -> 2x cvt_pk -> b64 store.
// ---------------------------------------------------------------------------
__global__ __launch_bounds__(256) void k_adjh(const float* __restrict__ adj,
                                              _Float16* __restrict__ adjh) {
  int idx = (blockIdx.x * 256 + threadIdx.x) * 4;  // element in (B,H,N) space
  if (idx >= B_ * H_ * N_) return;
  int bi = idx / N_;            // b*H + i'
  int j  = idx - bi * N_;
  int b  = bi / H_;
  int ii = bi - b * H_;
  float4 v = *(const float4*)(adj + ((size_t)b * N_ + V_ + ii) * N_ + j);
  union { _Float16 h[4]; float2 f2; } u;
  u.h[0] = (_Float16)v.x; u.h[1] = (_Float16)v.y;
  u.h[2] = (_Float16)v.z; u.h[3] = (_Float16)v.w;
  *(float2*)(adjh + idx) = u.f2;
}

// ---------------------------------------------------------------------------
// Kernel 1: x -> relu(.w_in1) -> relu(.w_in2) -> .w_e1[HD:]  => bvec (B,N,HD)
// Hidden rows (j >= V) are written as zeros. Grid: (B*N)/16 x 128 threads.
// ---------------------------------------------------------------------------
__global__ __launch_bounds__(128) void k_proj(
    const float* __restrict__ x, const _Float16* __restrict__ wT_in1,
    const float* __restrict__ b_in1, const _Float16* __restrict__ wT_in2,
    const float* __restrict__ b_in2, const _Float16* __restrict__ wT_e1s,
    float* __restrict__ bvec) {
  __shared__ __align__(16) _Float16 h1[16][HD_];
  __shared__ __align__(16) _Float16 h2[16][HD_];

  int row0 = blockIdx.x * 16;  // tile base in (B,N) row space
  int b    = row0 / N_;
  int jn0  = row0 % N_;
  int t    = threadIdx.x;

  if (jn0 >= V_) {  // hidden node rows: bvec = 0 (uniform branch)
    float* dst = bvec + (size_t)row0 * HD_;
    for (int idx = t; idx < 16 * HD_; idx += 128) dst[idx] = 0.f;
    return;
  }

  int col0 = (t >> 5) * 16;
  const float* xrow = x + ((size_t)b * V_ + jn0) * TD_;

  v8f c1 = {};
#pragma unroll
  for (int kt = 0; kt < TD_ / 32; ++kt)
    c1 = wmma_f16(a_frag_g32(xrow, TD_, kt * 32),
                  b_frag_t(wT_in1, TD_, kt * 32, col0), c1);
  c1 = bias_relu(c1, b_in1, col0);
  store_tile_lds<HD_>(&h1[0][0], col0, c1);
  __syncthreads();

  v8f c2 = {};
#pragma unroll
  for (int kt = 0; kt < HD_ / 32; ++kt)
    c2 = wmma_f16(a_frag_h16(&h1[0][0], HD_, kt * 32),
                  b_frag_t(wT_in2, HD_, kt * 32, col0), c2);
  c2 = bias_relu(c2, b_in2, col0);
  store_tile_lds<HD_>(&h2[0][0], col0, c2);
  __syncthreads();

  v8f c3 = {};  // sender projection: no bias, no relu
#pragma unroll
  for (int kt = 0; kt < HD_ / 32; ++kt)
    c3 = wmma_f16(a_frag_h16(&h2[0][0], HD_, kt * 32),
                  b_frag_t(wT_e1s, HD_, kt * 32, col0), c3);
  store_tile_g(bvec + (size_t)row0 * HD_, HD_, col0, c3);
}

// ---------------------------------------------------------------------------
// Kernel 2: F[b,j,:] = relu( relu(bvec[b,j,:]+b_e1) @ w_e2 + b_e2 ), stored
// TRANSPOSED as f16: Ft[b][h][j] (HD x N per batch). Grid: (B*N)/16 x 128.
// ---------------------------------------------------------------------------
__global__ __launch_bounds__(128) void k_edge(
    const float* __restrict__ bvec, const float* __restrict__ b_e1,
    const _Float16* __restrict__ wT_e2, const float* __restrict__ b_e2,
    _Float16* __restrict__ Ft) {
  __shared__ __align__(16) _Float16 es[16][HD_];
  int row0 = blockIdx.x * 16;
  int b    = row0 / N_;
  int jn0  = row0 % N_;
  int t    = threadIdx.x;

  const float* src = bvec + (size_t)row0 * HD_;
  for (int idx = t; idx < 16 * HD_; idx += 128) {
    int k = idx & (HD_ - 1);
    es[idx >> 6][k] = (_Float16)fmaxf(src[idx] + b_e1[k], 0.f);
  }
  __syncthreads();

  int col0 = (t >> 5) * 16;
  v8f c = {};
#pragma unroll
  for (int kt = 0; kt < HD_ / 32; ++kt)
    c = wmma_f16(a_frag_h16(&es[0][0], HD_, kt * 32),
                 b_frag_t(wT_e2, HD_, kt * 32, col0), c);
  c = bias_relu(c, b_e2, col0);

  // transposed store: Ft[b][col][jn0+row]
  _Float16* Fb = Ft + (size_t)b * HD_ * N_;
  int lane = t & 31;
  int col  = col0 + (lane & 15);
  int r0   = (lane >> 4) * 8;
#pragma unroll
  for (int r = 0; r < 8; ++r)
    Fb[(size_t)col * N_ + jn0 + r0 + r] = (_Float16)c[r];
}

// ---------------------------------------------------------------------------
// Kernel 3: agg = adjh[b] @ F[b]  (K=512, both operands f16, pure b128
// streams), then fused node MLP + output projection.
// Grid: (B*H)/16 = 32 blocks x 128 threads.
// ---------------------------------------------------------------------------
__global__ __launch_bounds__(128) void k_agg_out(
    const _Float16* __restrict__ adjh, const _Float16* __restrict__ Ft,
    const _Float16* __restrict__ wT_n1, const float* __restrict__ b_n1,
    const _Float16* __restrict__ wT_n2, const float* __restrict__ b_n2,
    const _Float16* __restrict__ wT_out, const float* __restrict__ b_out,
    float* __restrict__ out) {
  __shared__ __align__(16) _Float16 aggh[16][HD_];
  __shared__ __align__(16) _Float16 n1h[16][HD_];
  __shared__ __align__(16) _Float16 n2h[16][HD_];

  int g    = blockIdx.x;
  int b    = g >> 3;             // H/16 = 8 row-tiles per batch
  int it0  = (g & 7) * 16;       // i-tile offset inside hidden block
  int t    = threadIdx.x;
  int col0 = (t >> 5) * 16;

  const _Float16* adjb = adjh + (size_t)(b * H_ + it0) * N_;
  const _Float16* Fb   = Ft + (size_t)b * HD_ * N_;

  v8f acc = {};
#pragma unroll 4
  for (int kt = 0; kt < N_ / 32; ++kt)
    acc = wmma_f16(a_frag_h16(adjb, N_, kt * 32),
                   b_frag_t(Fb, N_, kt * 32, col0), acc);

  store_tile_lds<HD_>(&aggh[0][0], col0, acc);  // agg, no activation
  __syncthreads();

  v8f c1 = {};
#pragma unroll
  for (int kt = 0; kt < HD_ / 32; ++kt)
    c1 = wmma_f16(a_frag_h16(&aggh[0][0], HD_, kt * 32),
                  b_frag_t(wT_n1, HD_, kt * 32, col0), c1);
  c1 = bias_relu(c1, b_n1, col0);
  store_tile_lds<HD_>(&n1h[0][0], col0, c1);
  __syncthreads();

  v8f c2 = {};
#pragma unroll
  for (int kt = 0; kt < HD_ / 32; ++kt)
    c2 = wmma_f16(a_frag_h16(&n1h[0][0], HD_, kt * 32),
                  b_frag_t(wT_n2, HD_, kt * 32, col0), c2);
  c2 = bias_relu(c2, b_n2, col0);
  store_tile_lds<HD_>(&n2h[0][0], col0, c2);
  __syncthreads();

  // out_proj: 128 output cols; wave w owns col-tiles 2w and 2w+1.
  float* outb = out + ((size_t)b * H_ + it0) * TD_;
#pragma unroll
  for (int ct = 0; ct < 2; ++ct) {
    int oc0 = ((t >> 5) * 2 + ct) * 16;
    v8f co = {};
#pragma unroll
    for (int kt = 0; kt < HD_ / 32; ++kt)
      co = wmma_f16(a_frag_h16(&n2h[0][0], HD_, kt * 32),
                    b_frag_t(wT_out, HD_, kt * 32, oc0), co);
    int lane = t & 31;
    float bv = b_out[oc0 + (lane & 15)];
#pragma unroll
    for (int r = 0; r < 8; ++r) co[r] += bv;
    store_tile_g(outb, TD_, oc0, co);
  }
}

extern "C" void kernel_launch(void* const* d_in, const int* in_sizes, int n_in,
                              void* d_out, int out_size, void* d_ws,
                              size_t ws_size, hipStream_t stream) {
  const float* x     = (const float*)d_in[0];
  const float* adj   = (const float*)d_in[1];
  const float* w_in1 = (const float*)d_in[2];
  const float* b_in1 = (const float*)d_in[3];
  const float* w_in2 = (const float*)d_in[4];
  const float* b_in2 = (const float*)d_in[5];
  const float* w_e1  = (const float*)d_in[6];
  const float* b_e1  = (const float*)d_in[7];
  const float* w_e2  = (const float*)d_in[8];
  const float* b_e2  = (const float*)d_in[9];
  const float* w_n1  = (const float*)d_in[10];
  const float* b_n1  = (const float*)d_in[11];
  const float* w_n2  = (const float*)d_in[12];
  const float* b_n2  = (const float*)d_in[13];
  const float* w_out = (const float*)d_in[14];
  const float* b_out = (const float*)d_in[15];
  float* out = (float*)d_out;

  // Workspace layout (f16 region first, then f32 bvec; ~1.35 MB total):
  _Float16* wT_in1 = (_Float16*)d_ws;                 // 64 x 128
  _Float16* wT_in2 = wT_in1 + 64 * 128;               // 64 x 64
  _Float16* wT_e1s = wT_in2 + 64 * 64;                // 64 x 64
  _Float16* wT_e2  = wT_e1s + 64 * 64;                // 64 x 64
  _Float16* wT_n1  = wT_e2 + 64 * 64;                 // 64 x 64
  _Float16* wT_n2  = wT_n1 + 64 * 64;                 // 64 x 64
  _Float16* wT_out = wT_n2 + 64 * 64;                 // 128 x 64
  _Float16* Ft     = wT_out + 128 * 64;               // B * HD * N (f16)
  _Float16* adjh   = Ft + (size_t)B_ * HD_ * N_;      // B * H * N (f16)
  float*    bvec   = (float*)(adjh + (size_t)B_ * H_ * N_);  // B*N*HD f32

  auto tr = [&](const float* W, _Float16* Wt, int K, int C) {
    k_wt<<<(K * C + 255) / 256, 256, 0, stream>>>(W, Wt, K, C);
  };
  tr(w_in1, wT_in1, TD_, HD_);
  tr(w_in2, wT_in2, HD_, HD_);
  tr(w_e1 + HD_ * HD_, wT_e1s, HD_, HD_);  // sender half of w_e1
  tr(w_e2, wT_e2, HD_, HD_);
  tr(w_n1, wT_n1, HD_, HD_);
  tr(w_n2, wT_n2, HD_, HD_);
  tr(w_out, wT_out, HD_, TD_);
  k_adjh<<<(B_ * H_ * N_ / 4 + 255) / 256, 256, 0, stream>>>(adj, adjh);

  k_proj<<<(B_ * N_) / 16, 128, 0, stream>>>(x, wT_in1, b_in1, wT_in2, b_in2,
                                             wT_e1s, bvec);
  k_edge<<<(B_ * N_) / 16, 128, 0, stream>>>(bvec, b_e1, wT_e2, b_e2, Ft);
  k_agg_out<<<(B_ * H_) / 16, 128, 0, stream>>>(adjh, Ft, wT_n1, b_n1, wT_n2,
                                                b_n2, wT_out, b_out, out);
}
